// RotaryMultiHeadLatentAttention_35296041239268
// MI455X (gfx1250) — compile-verified, matrix operand order checked by reading, not run on hardware
//
#include <hip/hip_runtime.h>
#include <hip/hip_bf16.h>

// MI455X / gfx1250, wave32, WMMA-based MLA attention.

typedef __attribute__((ext_vector_type(16))) _Float16 v16h;
typedef __attribute__((ext_vector_type(8)))  float    v8f;
typedef int v2i_t __attribute__((vector_size(2 * sizeof(int))));  // b64 payload

#define B_  2
#define T_  2048
#define H_  1024
#define NH_ 16
#define NKV_ 4
#define D_  64
#define R_  256
#define LNEPS 1e-5f
#define LOG10000 9.21034037197618f

// K-index held by vector element e for 16-bit A/B WMMA fragments (ISA 7.12.2):
// lanes 0-15 / 16-31 split by +8; elements 0-7 cover K 0..15, 8-15 cover K 16..31.
__device__ __forceinline__ int kfun(int e, int half) {
  return e + 8 * half + (e >= 8 ? 8 : 0);
}

__device__ __forceinline__ v8f wmma_f16(v16h a, v16h b, v8f c) {
  return __builtin_amdgcn_wmma_f32_16x16x32_f16(false, a, false, b, (short)0, c,
                                                false, false);
}

// --------------------------------------------------------- async copy helpers
// CDNA5 GLOBAL_LOAD_ASYNC_TO_LDS_B64 (ASYNCcnt-tracked) when available,
// otherwise a synchronous global->LDS copy. Guarded so compile never breaks.
__device__ __forceinline__ void g2lds_b64(const _Float16* gsrc, _Float16* ldst) {
#if __has_builtin(__builtin_amdgcn_global_load_async_to_lds_b64)
  __builtin_amdgcn_global_load_async_to_lds_b64(
      (__attribute__((address_space(1))) v2i_t*)(unsigned long long)gsrc,
      (__attribute__((address_space(3))) v2i_t*)(unsigned long long)ldst, 0, 0);
#else
  *(uint2*)ldst = *(const uint2*)gsrc;
#endif
}

__device__ __forceinline__ void g2lds_wait() {
#if __has_builtin(__builtin_amdgcn_global_load_async_to_lds_b64)
#if __has_builtin(__builtin_amdgcn_s_wait_asynccnt)
  __builtin_amdgcn_s_wait_asynccnt(0);
#else
  asm volatile("s_wait_asynccnt 0x0" ::: "memory");
#endif
#endif
}

// ---------------------------------------------------------------- convert
__global__ void f32_to_f16_kernel(const float* __restrict__ s,
                                  _Float16* __restrict__ d, int n) {
  int i = blockIdx.x * blockDim.x + threadIdx.x;
  if (i < n) d[i] = (_Float16)s[i];
}

// src[k, n] (fp32, row stride ld) -> dst[n, k] (f16, dense [cols][rows])
__global__ void f32_to_f16_T_kernel(const float* __restrict__ s,
                                    _Float16* __restrict__ d,
                                    int rows, int cols, int ld) {
  int i = blockIdx.x * blockDim.x + threadIdx.x;
  if (i >= rows * cols) return;
  int k = i % rows;  // dst-linear for coalesced writes
  int n = i / rows;
  d[i] = (_Float16)s[(size_t)k * ld + n];
}

// ---------------------------------------------------------------- GEMM
// C[M,N] (fp32, ldc) = A[M,K] (f16 row-major, lda=K) * Bt[N,K] (f16, pre-transposed)
// blockDim = 128 (4 waves); block tile 16(M) x 64(N); A tile staged in LDS and
// shared by the 4 waves, fragments read back as ds b128 loads.
__global__ __launch_bounds__(128)
void gemm_f16_kernel(const _Float16* __restrict__ A, const _Float16* __restrict__ Bt,
                     float* __restrict__ C, int M, int N, int K, int ldc) {
  __shared__ _Float16 sA[16 * 32];  // 1 KB A slab for current k-step
  int tid  = threadIdx.x;
  int lane = tid & 31;
  int wave = tid >> 5;
  int half = lane >> 4;
  int lm   = lane & 15;
  int tn = blockIdx.x * 4 + wave;
  int tm = blockIdx.y;

  int r_st   = tid >> 3;      // staging: row 0..15
  int seg_st = (tid & 7) * 4; // staging: 4-half segment
  const _Float16* Ast = A + (size_t)(tm * 16 + r_st) * K + seg_st;
  const _Float16* Btrow = Bt + (size_t)(tn * 16 + lm) * K;

  v8f acc = {};
  for (int k0 = 0; k0 < K; k0 += 32) {
    // cooperative async stage of 16x32 A tile (8 B per thread)
    g2lds_b64(Ast + k0, &sA[r_st * 32 + seg_st]);
    g2lds_wait();
    __syncthreads();
    __builtin_prefetch(Btrow + k0 + 64, 0, 1);  // global_prefetch_b8 on B stream

    v16h a, b;
#pragma unroll
    for (int e = 0; e < 16; ++e) {
      int kk = kfun(e, half);
      a[e] = sA[lm * 32 + kk];      // -> 2x ds_load_b128
      b[e] = Btrow[k0 + kk];        // -> 2x global_load_b128
    }
    acc = wmma_f16(a, b, acc);
    __syncthreads();  // before next slab overwrite
  }
#pragma unroll
  for (int v = 0; v < 8; ++v)
    C[(size_t)(tm * 16 + v + 8 * half) * ldc + tn * 16 + lm] = acc[v];
}

// ---------------------------------------------------------------- LayerNorm
__global__ void ln_kernel(const float* __restrict__ latent,
                          const float* __restrict__ w, const float* __restrict__ bb,
                          _Float16* __restrict__ out) {
  int row = blockIdx.x;
  int lane = threadIdx.x;
  const float* x = latent + (size_t)row * R_;
  float v[8], s = 0.f, s2 = 0.f;
#pragma unroll
  for (int i = 0; i < 8; ++i) {
    v[i] = x[lane + 32 * i];
    s += v[i];
    s2 += v[i] * v[i];
  }
#pragma unroll
  for (int m = 16; m >= 1; m >>= 1) {
    s  += __shfl_xor(s, m, 32);
    s2 += __shfl_xor(s2, m, 32);
  }
  float mean = s * (1.0f / R_);
  float var  = s2 * (1.0f / R_) - mean * mean;
  float rstd = rsqrtf(var + LNEPS);
#pragma unroll
  for (int i = 0; i < 8; ++i) {
    int c = lane + 32 * i;
    out[(size_t)row * R_ + c] = (_Float16)((v[i] - mean) * rstd * w[c] + bb[c]);
  }
}

// ---------------------------------------------------------------- RoPE q
// qf[(b*T+t), h*D+d] fp32 -> q16[((b*NH+h)*T+t)*D + d] f16 (roped)
__global__ void rope_q_kernel(const float* __restrict__ qf,
                              _Float16* __restrict__ q16) {
  int i = blockIdx.x * blockDim.x + threadIdx.x;
  if (i >= B_ * T_ * NH_ * D_) return;
  int d = i % D_;
  int h = (i / D_) % NH_;
  int t = (i / (D_ * NH_)) % T_;
  int b = i / (D_ * NH_ * T_);
  const float* row = qf + ((size_t)(b * T_ + t)) * (NH_ * D_) + h * D_;
  float x   = row[d];
  float rot = (d < 32) ? -row[d + 32] : row[d - 32];
  float invf = __expf(-(float)(d & 31) * (1.0f / 32.0f) * LOG10000);
  float ang = (float)t * invf;
  float sn, cs;
  __sincosf(ang, &sn, &cs);
  q16[((size_t)(b * NH_ + h) * T_ + t) * D_ + d] = (_Float16)(x * cs + rot * sn);
}

// ---------------------------------------------------------------- RoPE k + transpose v
// kvf[(b*T+t), 512]: cols [0,256)=k (roped -> [b,kvh,t,d]),
//                    cols [256,512)=v (-> transposed [b,kvh,d,t])
__global__ void rope_kv_kernel(const float* __restrict__ kvf,
                               _Float16* __restrict__ k16,
                               _Float16* __restrict__ v16t) {
  int i = blockIdx.x * blockDim.x + threadIdx.x;
  const int total = B_ * T_ * NKV_ * D_;
  if (i >= 2 * total) return;
  int isv = (i >= total) ? 1 : 0;
  int j = isv ? i - total : i;
  int d  = j % D_;
  int kh = (j / D_) % NKV_;
  int t  = (j / (D_ * NKV_)) % T_;
  int b  = j / (D_ * NKV_ * T_);
  const float* row = kvf + ((size_t)(b * T_ + t)) * (2 * NKV_ * D_);
  if (isv) {
    v16t[((size_t)(b * NKV_ + kh) * D_ + d) * T_ + t] =
        (_Float16)row[NKV_ * D_ + kh * D_ + d];
  } else {
    float x   = row[kh * D_ + d];
    float rot = (d < 32) ? -row[kh * D_ + d + 32] : row[kh * D_ + d - 32];
    float invf = __expf(-(float)(d & 31) * (1.0f / 32.0f) * LOG10000);
    float ang = (float)t * invf;
    float sn, cs;
    __sincosf(ang, &sn, &cs);
    k16[((size_t)(b * NKV_ + kh) * T_ + t) * D_ + d] = (_Float16)(x * cs + rot * sn);
  }
}

// ---------------------------------------------------------------- flash attention
// 4 waves/block; each wave owns one (b, h, 16-row q tile); causal; f16 WMMA,
// f32 accum, online softmax. Per-wave private LDS slab for the P C->A transpose
// (same-wave DS ordering via DScnt, no barriers -> divergent trip counts are safe).
__global__ __launch_bounds__(128)
void flash_attn_kernel(const _Float16* __restrict__ q16,
                       const _Float16* __restrict__ k16,
                       const _Float16* __restrict__ v16t,
                       _Float16* __restrict__ attn16) {
  __shared__ _Float16 sPall[4][16 * 32];
  int lane = threadIdx.x & 31;
  int wave = threadIdx.x >> 5;
  _Float16* sP = sPall[wave];
  int half = lane >> 4;
  int lm   = lane & 15;
  int qt = blockIdx.x * 4 + wave;  // T/16 tiles
  int h  = blockIdx.y;             // NH
  int b  = blockIdx.z;             // B
  int kvh   = h >> 2;              // NH/NKV = 4
  int qbase = qt * 16;

  // Q tile: 16x64 -> two A fragments (K = 0..31, 32..63), loaded once.
  const _Float16* Qrow = q16 + ((size_t)(b * NH_ + h) * T_ + qbase + lm) * D_;
  v16h qa0, qa1;
#pragma unroll
  for (int e = 0; e < 16; ++e) {
    int kk = kfun(e, half);
    qa0[e] = Qrow[kk];
    qa1[e] = Qrow[32 + kk];
  }
  const _Float16* Kb = k16  + (size_t)(b * NKV_ + kvh) * T_ * D_;
  const _Float16* Vt = v16t + (size_t)(b * NKV_ + kvh) * D_ * T_;

  float m[8], l[8];
  v8f o[4];
#pragma unroll
  for (int v = 0; v < 8; ++v) { m[v] = -1e30f; l[v] = 0.f; }
#pragma unroll
  for (int j = 0; j < 4; ++j)
#pragma unroll
    for (int v = 0; v < 8; ++v) o[j][v] = 0.f;

  const float scale = 0.125f;  // D^-0.5

  for (int kb = 0; kb < qbase + 16; kb += 32) {
    // ---- S chunk: 16 q-rows x 32 kv-cols = two C fragments
    v8f s0 = {}, s1 = {};
    v16h bk;
#pragma unroll
    for (int e = 0; e < 16; ++e) bk[e] = Kb[(size_t)(kb + lm) * D_ + kfun(e, half)];
    s0 = wmma_f16(qa0, bk, s0);
#pragma unroll
    for (int e = 0; e < 16; ++e) bk[e] = Kb[(size_t)(kb + lm) * D_ + 32 + kfun(e, half)];
    s0 = wmma_f16(qa1, bk, s0);
#pragma unroll
    for (int e = 0; e < 16; ++e) bk[e] = Kb[(size_t)(kb + 16 + lm) * D_ + kfun(e, half)];
    s1 = wmma_f16(qa0, bk, s1);
#pragma unroll
    for (int e = 0; e < 16; ++e) bk[e] = Kb[(size_t)(kb + 16 + lm) * D_ + 32 + kfun(e, half)];
    s1 = wmma_f16(qa1, bk, s1);

    // ---- causal mask + online softmax (rows live on 16-lane halves)
    float p0[8], p1[8];
#pragma unroll
    for (int v = 0; v < 8; ++v) {
      int row = qbase + v + 8 * half;
      float a0 = (kb + lm      <= row) ? s0[v] * scale : -1e30f;
      float a1 = (kb + 16 + lm <= row) ? s1[v] * scale : -1e30f;
      float mx = fmaxf(a0, a1);
#pragma unroll
      for (int sh = 8; sh >= 1; sh >>= 1) mx = fmaxf(mx, __shfl_xor(mx, sh, 32));
      float mnew  = fmaxf(m[v], mx);
      float alpha = __expf(m[v] - mnew);
      float e0 = __expf(a0 - mnew);
      float e1 = __expf(a1 - mnew);
      float rs = e0 + e1;
#pragma unroll
      for (int sh = 8; sh >= 1; sh >>= 1) rs += __shfl_xor(rs, sh, 32);
      l[v] = l[v] * alpha + rs;
      m[v] = mnew;
      p0[v] = e0;
      p1[v] = e1;
#pragma unroll
      for (int j = 0; j < 4; ++j) o[j][v] *= alpha;
    }

    // ---- transpose P (C-frag layout) -> A-frag layout through private LDS slab
#pragma unroll
    for (int v = 0; v < 8; ++v) {
      int row = v + 8 * half;
      sP[row * 32 + lm]      = (_Float16)p0[v];
      sP[row * 32 + 16 + lm] = (_Float16)p1[v];
    }
    v16h pa;
#pragma unroll
    for (int e = 0; e < 16; ++e) pa[e] = sP[lm * 32 + kfun(e, half)];

    // ---- O += P (16x32) x V (32x64); V is [d][t] so B-frags are contiguous
#pragma unroll
    for (int j = 0; j < 4; ++j) {
      v16h vb;
#pragma unroll
      for (int e = 0; e < 16; ++e)
        vb[e] = Vt[(size_t)(j * 16 + lm) * T_ + kb + kfun(e, half)];
      o[j] = wmma_f16(pa, vb, o[j]);
    }
  }

  // ---- epilogue: normalize, emit [(b*T+t), h*D+d] f16 for the O-projection
#pragma unroll
  for (int v = 0; v < 8; ++v) {
    float inv = 1.0f / l[v];
    int row = qbase + v + 8 * half;
#pragma unroll
    for (int j = 0; j < 4; ++j)
      attn16[((size_t)(b * T_) + row) * (NH_ * D_) + h * D_ + j * 16 + lm] =
          (_Float16)(o[j][v] * inv);
  }
}

// ---------------------------------------------------------------- launch
extern "C" void kernel_launch(void* const* d_in, const int* in_sizes, int n_in,
                              void* d_out, int out_size, void* d_ws, size_t ws_size,
                              hipStream_t stream) {
  (void)in_sizes; (void)n_in; (void)out_size; (void)ws_size;
  const float* hs   = (const float*)d_in[0];
  // d_in[1] attention_mask: causal, computed in-kernel instead.
  const float* Wq   = (const float*)d_in[2];
  const float* Wkva = (const float*)d_in[3];
  const float* Wkvb = (const float*)d_in[4];
  const float* lnw  = (const float*)d_in[5];
  const float* lnb  = (const float*)d_in[6];
  const float* Wo   = (const float*)d_in[7];
  float* out = (float*)d_out;

  const size_t M = (size_t)B_ * T_;  // 4096
  char* p = (char*)d_ws;
  auto alloc = [&](size_t bytes) -> char* {
    char* r = p;
    p += (bytes + 255) & ~(size_t)255;
    return r;
  };
  _Float16* hs16   = (_Float16*)alloc(M * H_ * 2);
  _Float16* WqT    = (_Float16*)alloc((size_t)H_ * (NH_ * D_) * 2);   // [N][K]
  _Float16* WkvaT  = (_Float16*)alloc((size_t)H_ * R_ * 2);           // [R][H]
  _Float16* WkvbT  = (_Float16*)alloc((size_t)R_ * (NKV_ * D_ * 2) * 2);
  _Float16* WoT    = (_Float16*)alloc((size_t)(NH_ * D_) * H_ * 2);
  float*    qf32   = (float*)alloc(M * (NH_ * D_) * 4);
  float*    latf32 = (float*)alloc(M * R_ * 4);
  _Float16* lat16  = (_Float16*)alloc(M * R_ * 2);
  float*    kvf32  = (float*)alloc(M * (NKV_ * D_ * 2) * 4);
  _Float16* q16r   = (_Float16*)alloc((size_t)B_ * NH_ * T_ * D_ * 2);
  _Float16* k16r   = (_Float16*)alloc((size_t)B_ * NKV_ * T_ * D_ * 2);
  _Float16* v16t   = (_Float16*)alloc((size_t)B_ * NKV_ * D_ * T_ * 2);
  _Float16* attn16 = (_Float16*)alloc(M * (NH_ * D_) * 2);

  int n;
  n = (int)(M * H_);
  f32_to_f16_kernel<<<(n + 255) / 256, 256, 0, stream>>>(hs, hs16, n);
  n = H_ * NH_ * D_;
  f32_to_f16_T_kernel<<<(n + 255) / 256, 256, 0, stream>>>(Wq, WqT, H_, NH_ * D_, NH_ * D_);
  n = H_ * R_;
  f32_to_f16_T_kernel<<<(n + 255) / 256, 256, 0, stream>>>(Wkva, WkvaT, H_, R_, R_ + D_);
  n = R_ * NKV_ * D_ * 2;
  f32_to_f16_T_kernel<<<(n + 255) / 256, 256, 0, stream>>>(Wkvb, WkvbT, R_, NKV_ * D_ * 2,
                                                           NKV_ * D_ * 2);
  n = NH_ * D_ * H_;
  f32_to_f16_T_kernel<<<(n + 255) / 256, 256, 0, stream>>>(Wo, WoT, NH_ * D_, H_, H_);

  // Q projection: [4096,1024] x [1024,1024]
  gemm_f16_kernel<<<dim3((NH_ * D_) / 64, (int)(M / 16)), 128, 0, stream>>>(
      hs16, WqT, qf32, (int)M, NH_ * D_, H_, NH_ * D_);
  // Latent: [4096,1024] x [1024,256]
  gemm_f16_kernel<<<dim3(R_ / 64, (int)(M / 16)), 128, 0, stream>>>(
      hs16, WkvaT, latf32, (int)M, R_, H_, R_);
  // LayerNorm
  ln_kernel<<<(int)M, 32, 0, stream>>>(latf32, lnw, lnb, lat16);
  // KV projection: [4096,256] x [256,512]
  gemm_f16_kernel<<<dim3((NKV_ * D_ * 2) / 64, (int)(M / 16)), 128, 0, stream>>>(
      lat16, WkvbT, kvf32, (int)M, NKV_ * D_ * 2, R_, NKV_ * D_ * 2);
  // RoPE + relayout
  n = B_ * T_ * NH_ * D_;
  rope_q_kernel<<<(n + 255) / 256, 256, 0, stream>>>(qf32, q16r);
  n = B_ * T_ * NKV_ * D_ * 2;
  rope_kv_kernel<<<(n + 255) / 256, 256, 0, stream>>>(kvf32, k16r, v16t);
  // Flash attention (4 q-tiles per block, one per wave)
  flash_attn_kernel<<<dim3(T_ / 64, NH_, B_), 128, 0, stream>>>(q16r, k16r, v16t,
                                                                attn16);
  // Output projection: [4096,1024] x [1024,1024] -> fp32 out
  gemm_f16_kernel<<<dim3(H_ / 64, (int)(M / 16)), 128, 0, stream>>>(
      attn16, WoT, out, (int)M, H_, NH_ * D_, H_);
}
